// NetworkFlowGCN_79499844649437
// MI455X (gfx1250) — matrix operand order repaired and checked by reading, not compiled, not run on hardware
//
#include <hip/hip_runtime.h>

// ---------------------------------------------------------------------------
// Types for WMMA
// ---------------------------------------------------------------------------
typedef __attribute__((ext_vector_type(16))) _Float16 v16h;
typedef __attribute__((ext_vector_type(8)))  float    v8f;

// ---------------------------------------------------------------------------
// Small utility kernels
// ---------------------------------------------------------------------------
__global__ void fill_kernel(float* __restrict__ p, float v, size_t n) {
    size_t i = (size_t)blockIdx.x * blockDim.x + threadIdx.x;
    if (i < n) p[i] = v;
}

__global__ void deg_accum_kernel(const int* __restrict__ dst, float* __restrict__ deg, int E) {
    size_t i = (size_t)blockIdx.x * blockDim.x + threadIdx.x;
    if (i < (size_t)E) atomicAdd(&deg[dst[i]], 1.0f);
}

__global__ void rsqrt_kernel(float* __restrict__ p, size_t n) {
    size_t i = (size_t)blockIdx.x * blockDim.x + threadIdx.x;
    if (i < n) p[i] = rsqrtf(p[i]);
}

__global__ void cvt_f16_kernel(const float* __restrict__ in, _Float16* __restrict__ out, size_t n) {
    size_t i = (size_t)blockIdx.x * blockDim.x + threadIdx.x;
    if (i < n) out[i] = (_Float16)in[i];
}

// Pack weight matrix W[K x Kout] (f32, row-major) into per-wave WMMA B fragments.
// B order used here (contiguous k, same as A) for v_wmma_f32_16x16x32_f16: lane l holds column (l&15),
// K indices (l>>4)*16 + e for fragment element e in [0,16).
__global__ void pack_w_kernel(const float* __restrict__ W, _Float16* __restrict__ Wp,
                              int K, int Kout) {
    int idx = blockIdx.x * blockDim.x + threadIdx.x;
    int total = K * Kout;
    if (idx >= total) return;
    int CT   = Kout >> 4;
    int e    = idx & 15;
    int lane = (idx >> 4) & 31;
    int tile = idx >> 9;
    int ct   = tile % CT;
    int kt   = tile / CT;
    int col  = ct * 16 + (lane & 15);
    int k    = kt * 32 + ((lane >> 4) << 4) + e;
    Wp[idx] = (_Float16)W[(size_t)k * Kout + col];
}

// ---------------------------------------------------------------------------
// WMMA GEMM: C[N x Kout] = A[N x K] (f16) * Wp (packed f16) (+bias, +relu)
// One wave per 16x16 output tile; up to 8 waves/block (min(CT,8)) share the same 16 A rows.
// ---------------------------------------------------------------------------
__global__ void gemm_wmma_kernel(const _Float16* __restrict__ A,
                                 const _Float16* __restrict__ Wp,
                                 float* __restrict__ C,
                                 const float* __restrict__ bias,
                                 int N, int K, int Kout, int relu) {
    const int lane    = threadIdx.x;        // 0..31
    const int wave    = threadIdx.y;        // 0..NW-1
    const int NW      = blockDim.y;
    const int rowBase = blockIdx.x * 16;
    const int ct      = blockIdx.y * NW + wave;
    const int colBase = ct * 16;
    const int CT      = Kout >> 4;
    const int KT      = K >> 5;

    const int r     = lane & 15;
    const int khalf = (lane >> 4) << 4;     // 0 or 16 (A fragment K sub-offset)

    v8f acc = {};
    const _Float16* arow = A + (size_t)(rowBase + r) * K;

    for (int kt = 0; kt < KT; ++kt) {
        const int k0 = kt << 5;
        // A fragment: elements 0..15 = K[k0+khalf .. +15]
        v16h a;
        ((float4*)&a)[0] = *(const float4*)(arow + k0 + khalf);
        ((float4*)&a)[1] = *(const float4*)(arow + k0 + khalf + 8);
        // B fragment: packed contiguous (32 bytes per lane, fully coalesced)
        v16h b;
        const float4* pb = (const float4*)(Wp + (((size_t)(kt * CT + ct) * 32 + lane) << 4));
        ((float4*)&b)[0] = pb[0];
        ((float4*)&b)[1] = pb[1];

        acc = __builtin_amdgcn_wmma_f32_16x16x32_f16(
            /*neg_a=*/false, a, /*neg_b=*/false, b,
            /*c_mod=*/(short)0, acc, /*reuse_a=*/false, /*reuse_b=*/false);
    }

    const int col = colBase + (lane & 15);
    const float bv = bias ? bias[col] : 0.0f;
    const int rhi = (lane >> 4) << 3;       // 0 or 8
    #pragma unroll
    for (int v = 0; v < 8; ++v) {
        int row = rowBase + v + rhi;
        float o = acc[v] + bv;
        if (relu) o = fmaxf(o, 0.0f);
        C[(size_t)row * Kout + col] = o;
    }
}

// ---------------------------------------------------------------------------
// GCN aggregation: out[dst] += h[src] * dinv[src]*dinv[dst] over E edges +
// N self-loops. One thread per (edge, 4-float chunk).
// ---------------------------------------------------------------------------
__global__ void gcn_agg_kernel(const float* __restrict__ h,
                               const int* __restrict__ src,
                               const int* __restrict__ dst,
                               const float* __restrict__ dinv,
                               float* __restrict__ out,
                               int E, int N, int dim, int qshift) {
    size_t idx = (size_t)blockIdx.x * blockDim.x + threadIdx.x;
    size_t total = (size_t)(E + N) << qshift;
    if (idx >= total) return;
    size_t e = idx >> qshift;
    int q = (int)(idx & (((size_t)1 << qshift) - 1));
    int s, d;
    if (e < (size_t)E) { s = src[e]; d = dst[e]; }
    else               { s = d = (int)(e - (size_t)E); }
    float nrm = dinv[s] * dinv[d];
    float4 v = *(const float4*)(h + (size_t)s * dim + q * 4);
    float* o = out + (size_t)d * dim + q * 4;
    atomicAdd(o + 0, v.x * nrm);
    atomicAdd(o + 1, v.y * nrm);
    atomicAdd(o + 2, v.z * nrm);
    atomicAdd(o + 3, v.w * nrm);
}

// ---------------------------------------------------------------------------
// BatchNorm: per-column sum / sum-of-squares with block-partial reduction
// ---------------------------------------------------------------------------
#define BN_ROWS 128
__global__ void bn_stats_kernel(const float* __restrict__ x, float* __restrict__ s1,
                                float* __restrict__ s2, int N, int dim) {
    int col = threadIdx.x;           // blockDim.x == dim
    if (col >= dim) return;
    int r0 = blockIdx.x * BN_ROWS;
    int r1 = r0 + BN_ROWS; if (r1 > N) r1 = N;
    float a = 0.0f, b = 0.0f;
    for (int r = r0; r < r1; ++r) {
        float v = x[(size_t)r * dim + col];
        a += v; b += v * v;
    }
    atomicAdd(&s1[col], a);
    atomicAdd(&s2[col], b);
}

// Normalize + affine + ReLU, write f16 (feeds next WMMA GEMM directly)
__global__ void bn_apply_kernel(const float* __restrict__ x, const float* __restrict__ s1,
                                const float* __restrict__ s2, const float* __restrict__ g,
                                const float* __restrict__ be, _Float16* __restrict__ out,
                                size_t total, int dim, float invN) {
    size_t idx = (size_t)blockIdx.x * blockDim.x + threadIdx.x;
    if (idx >= total) return;
    int c = (int)(idx % (size_t)dim);
    float mean = s1[c] * invN;
    float var  = s2[c] * invN - mean * mean;
    float v = (x[idx] - mean) * rsqrtf(var + 1e-5f) * g[c] + be[c];
    out[idx] = (_Float16)fmaxf(v, 0.0f);
}

// ---------------------------------------------------------------------------
// GAT kernels
// ---------------------------------------------------------------------------
__global__ void gat_score_kernel(const float* __restrict__ h4,
                                 const float* __restrict__ a_src,
                                 const float* __restrict__ a_dst,
                                 float* __restrict__ es, float* __restrict__ ed, int N) {
    size_t idx = (size_t)blockIdx.x * blockDim.x + threadIdx.x;
    if (idx >= (size_t)N * 4) return;
    int hd = (int)(idx & 3);
    size_t n = idx >> 2;
    const float* hp = h4 + n * 256 + hd * 64;
    const float* as = a_src + hd * 64;
    const float* ad = a_dst + hd * 64;
    float s = 0.0f, d = 0.0f;
    #pragma unroll 8
    for (int c = 0; c < 64; ++c) { float v = hp[c]; s += v * as[c]; d += v * ad[c]; }
    es[idx] = s; ed[idx] = d;
}

__device__ __forceinline__ float lrelu02(float v) { return v > 0.0f ? v : 0.2f * v; }

__global__ void gat_max_kernel(const int* __restrict__ src, const int* __restrict__ dst,
                               const float* __restrict__ es, const float* __restrict__ ed,
                               float* __restrict__ emax, int E, int N) {
    size_t idx = (size_t)blockIdx.x * blockDim.x + threadIdx.x;
    size_t total = (size_t)(E + N) * 4;
    if (idx >= total) return;
    int hd = (int)(idx & 3);
    size_t e = idx >> 2;
    int s, d;
    if (e < (size_t)E) { s = src[e]; d = dst[e]; } else { s = d = (int)(e - (size_t)E); }
    float v = lrelu02(es[(size_t)s * 4 + hd] + ed[(size_t)d * 4 + hd]);
    atomicMax(&emax[(size_t)d * 4 + hd], v);   // global_atomic_max_num_f32
}

__global__ void gat_denom_kernel(const int* __restrict__ src, const int* __restrict__ dst,
                                 const float* __restrict__ es, const float* __restrict__ ed,
                                 const float* __restrict__ emax, float* __restrict__ denom,
                                 int E, int N) {
    size_t idx = (size_t)blockIdx.x * blockDim.x + threadIdx.x;
    size_t total = (size_t)(E + N) * 4;
    if (idx >= total) return;
    int hd = (int)(idx & 3);
    size_t e = idx >> 2;
    int s, d;
    if (e < (size_t)E) { s = src[e]; d = dst[e]; } else { s = d = (int)(e - (size_t)E); }
    float v = lrelu02(es[(size_t)s * 4 + hd] + ed[(size_t)d * 4 + hd]);
    atomicAdd(&denom[(size_t)d * 4 + hd], __expf(v - emax[(size_t)d * 4 + hd]));
}

// Weighted aggregation fused with head-mean: out[d][c] += 0.25 * sum_h alpha_h * h4[s][h][c]
__global__ void gat_agg_kernel(const int* __restrict__ src, const int* __restrict__ dst,
                               const float* __restrict__ h4,
                               const float* __restrict__ es, const float* __restrict__ ed,
                               const float* __restrict__ emax, const float* __restrict__ denom,
                               float* __restrict__ out, int E, int N) {
    size_t idx = (size_t)blockIdx.x * blockDim.x + threadIdx.x;
    size_t total = (size_t)(E + N) * 16;
    if (idx >= total) return;
    int q = (int)(idx & 15);
    size_t e = idx >> 4;
    int s, d;
    if (e < (size_t)E) { s = src[e]; d = dst[e]; } else { s = d = (int)(e - (size_t)E); }
    float4 acc = {0.0f, 0.0f, 0.0f, 0.0f};
    #pragma unroll
    for (int hd = 0; hd < 4; ++hd) {
        float v = lrelu02(es[(size_t)s * 4 + hd] + ed[(size_t)d * 4 + hd]);
        float a = __expf(v - emax[(size_t)d * 4 + hd]) / denom[(size_t)d * 4 + hd];
        float4 hv = *(const float4*)(h4 + (size_t)s * 256 + hd * 64 + q * 4);
        acc.x += a * hv.x; acc.y += a * hv.y; acc.z += a * hv.z; acc.w += a * hv.w;
    }
    float* o = out + (size_t)d * 64 + q * 4;
    atomicAdd(o + 0, 0.25f * acc.x);
    atomicAdd(o + 1, 0.25f * acc.y);
    atomicAdd(o + 2, 0.25f * acc.z);
    atomicAdd(o + 3, 0.25f * acc.w);
}

__global__ void add_bias_f16_kernel(const float* __restrict__ in, const float* __restrict__ bias,
                                    _Float16* __restrict__ out, size_t total) {
    size_t idx = (size_t)blockIdx.x * blockDim.x + threadIdx.x;
    if (idx >= total) return;
    out[idx] = (_Float16)(in[idx] + bias[idx & 63]);
}

// Final tiny classifier: logits = h[Nx32] @ Wc2[32x6] + bc2
__global__ void final_kernel(const float* __restrict__ h, const float* __restrict__ Wc2,
                             const float* __restrict__ bc2, float* __restrict__ out, int N) {
    int n = blockIdx.x * blockDim.x + threadIdx.x;
    if (n >= N) return;
    float acc[6];
    #pragma unroll
    for (int c = 0; c < 6; ++c) acc[c] = bc2[c];
    const float* hp = h + (size_t)n * 32;
    for (int k = 0; k < 32; ++k) {
        float v = hp[k];
        #pragma unroll
        for (int c = 0; c < 6; ++c) acc[c] += v * Wc2[k * 6 + c];
    }
    #pragma unroll
    for (int c = 0; c < 6; ++c) out[(size_t)n * 6 + c] = acc[c];
}

// ---------------------------------------------------------------------------
// Host orchestration
// ---------------------------------------------------------------------------
static inline unsigned cdivu(size_t a, size_t b) { return (unsigned)((a + b - 1) / b); }

extern "C" void kernel_launch(void* const* d_in, const int* in_sizes, int n_in,
                              void* d_out, int out_size, void* d_ws, size_t ws_size,
                              hipStream_t stream) {
    const float* x    = (const float*)d_in[0];
    const int*   ei   = (const int*)d_in[1];
    const float* W1   = (const float*)d_in[2];
    const float* g1   = (const float*)d_in[4];
    const float* be1  = (const float*)d_in[5];
    const float* W2   = (const float*)d_in[6];
    const float* g2   = (const float*)d_in[8];
    const float* be2  = (const float*)d_in[9];
    const float* W3   = (const float*)d_in[10];
    const float* g3   = (const float*)d_in[12];
    const float* be3  = (const float*)d_in[13];
    const float* Wg   = (const float*)d_in[14];
    const float* a_s  = (const float*)d_in[15];
    const float* a_d  = (const float*)d_in[16];
    const float* bg   = (const float*)d_in[17];
    const float* Wc1  = (const float*)d_in[18];
    const float* bc1  = (const float*)d_in[19];
    const float* Wc2  = (const float*)d_in[20];
    const float* bc2  = (const float*)d_in[21];

    const int N = in_sizes[0] / 128;
    const int E = in_sizes[1] / 2;
    const int* srcA = ei;
    const int* dstA = ei + E;

    // Workspace carve-up (all offsets keep 16B alignment)
    float*     ws    = (float*)d_ws;
    float*     deg   = ws;                                   // N (becomes dinv)
    float*     s1    = deg + N;                              // 256
    float*     s2    = s1 + 256;                             // 256
    float*     bufA  = s2 + 256;                             // N*256  (GEMM outputs / GAT h4)
    float*     bufB  = bufA + (size_t)N * 256;               // N*256  (aggregation target)
    _Float16*  h16   = (_Float16*)(bufB + (size_t)N * 256);  // N*256 halfs (GEMM A input)
    _Float16*  wpk   = h16 + (size_t)N * 256;                // 256*256 halfs (packed weights)
    float*     es    = (float*)(wpk + 256 * 256);            // N*4
    float*     ed    = es + (size_t)N * 4;                   // N*4
    float*     emax  = ed + (size_t)N * 4;                   // N*4
    float*     den   = emax + (size_t)N * 4;                 // N*4
    float*     out   = (float*)d_out;

    const float invN = 1.0f / (float)N;

    // ---- degrees: deg = 1 (self loop) + indeg; dinv = rsqrt(deg)
    fill_kernel<<<cdivu(N, 256), 256, 0, stream>>>(deg, 1.0f, (size_t)N);
    deg_accum_kernel<<<cdivu(E, 256), 256, 0, stream>>>(dstA, deg, E);
    rsqrt_kernel<<<cdivu(N, 256), 256, 0, stream>>>(deg, (size_t)N);

    auto run_gemm = [&](const _Float16* Ain, const float* W, int K, int Kout,
                        float* Cout, const float* bias, int relu) {
        pack_w_kernel<<<cdivu((size_t)K * Kout, 256), 256, 0, stream>>>(W, wpk, K, Kout);
        int CT = Kout / 16;
        int NW = CT < 8 ? CT : 8;
        dim3 blk(32, NW), grd(N / 16, CT / NW);
        gemm_wmma_kernel<<<grd, blk, 0, stream>>>(Ain, wpk, Cout, bias, N, K, Kout, relu);
    };

    auto run_gcn_bn = [&](int K, int Kout, const float* W, const float* g, const float* be) {
        run_gemm(h16, W, K, Kout, bufA, nullptr, 0);
        fill_kernel<<<cdivu((size_t)N * Kout, 256), 256, 0, stream>>>(bufB, 0.0f, (size_t)N * Kout);
        int qshift = (Kout == 256) ? 6 : (Kout == 128) ? 5 : 4;   // log2(Kout/4)
        size_t tot = (size_t)(E + N) << qshift;
        gcn_agg_kernel<<<cdivu(tot, 256), 256, 0, stream>>>(bufA, srcA, dstA, deg, bufB,
                                                            E, N, Kout, qshift);
        fill_kernel<<<2, 256, 0, stream>>>(s1, 0.0f, (size_t)512);  // zeros s1 and s2
        bn_stats_kernel<<<cdivu(N, BN_ROWS), Kout, 0, stream>>>(bufB, s1, s2, N, Kout);
        bn_apply_kernel<<<cdivu((size_t)N * Kout, 256), 256, 0, stream>>>(
            bufB, s1, s2, g, be, h16, (size_t)N * Kout, Kout, invN);
    };

    // ---- GCN stack (bias folded away by the following BatchNorm)
    cvt_f16_kernel<<<cdivu((size_t)N * 128, 256), 256, 0, stream>>>(x, h16, (size_t)N * 128);
    run_gcn_bn(128, 256, W1, g1, be1);
    run_gcn_bn(256, 128, W2, g2, be2);
    run_gcn_bn(128, 64,  W3, g3, be3);

    // ---- GAT
    run_gemm(h16, Wg, 64, 256, bufA, nullptr, 0);   // h4 = h @ Wg  [N,4,64]
    gat_score_kernel<<<cdivu((size_t)N * 4, 256), 256, 0, stream>>>(bufA, a_s, a_d, es, ed, N);
    fill_kernel<<<cdivu((size_t)N * 4, 256), 256, 0, stream>>>(emax, -3.0e38f, (size_t)N * 4);
    fill_kernel<<<cdivu((size_t)N * 4, 256), 256, 0, stream>>>(den, 0.0f, (size_t)N * 4);
    {
        size_t tot = (size_t)(E + N) * 4;
        gat_max_kernel<<<cdivu(tot, 256), 256, 0, stream>>>(srcA, dstA, es, ed, emax, E, N);
        gat_denom_kernel<<<cdivu(tot, 256), 256, 0, stream>>>(srcA, dstA, es, ed, emax, den, E, N);
    }
    fill_kernel<<<cdivu((size_t)N * 64, 256), 256, 0, stream>>>(bufB, 0.0f, (size_t)N * 64);
    {
        size_t tot = (size_t)(E + N) * 16;
        gat_agg_kernel<<<cdivu(tot, 256), 256, 0, stream>>>(srcA, dstA, bufA, es, ed, emax,
                                                            den, bufB, E, N);
    }
    add_bias_f16_kernel<<<cdivu((size_t)N * 64, 256), 256, 0, stream>>>(bufB, bg, h16,
                                                                        (size_t)N * 64);

    // ---- Classifier MLP
    run_gemm(h16, Wc1, 64, 32, bufA, bc1, 1);       // relu(h @ Wc1 + bc1) [N,32]
    final_kernel<<<cdivu(N, 256), 256, 0, stream>>>(bufA, Wc2, bc2, out, N);
}
